// ScannedRNN_88510686036629
// MI455X (gfx1250) — compile-verified, matrix operand order checked by reading, not run on hardware
//
#include <hip/hip_runtime.h>
#include <hip/hip_bf16.h>

// GRU scan, persistent-kernel design for MI455X (gfx1250, wave32, WMMA).
// T=1024, B=256, D=256, H=256.
// d_out = [final_carry (B*H) | ys (T*B*H)] fp32.
//
// Critical-path design: the 1024-step scan is latency bound. All
// time-invariant work (weight bf16 pack -> LDS B-layout, x bf16 conversion)
// is hoisted off the chain; per step each wave issues 48
// v_wmma_f32_16x16x32_bf16 + transcendental VALU + one device barrier.

#define T_STEPS 1024
#define BATCH   256
#define DIM     256
#define HID     256
#define NWG     16      // one workgroup per 16-wide H slice
#define THREADS 512     // 16 waves; wave w owns M-tile w (16 batch rows)

typedef __attribute__((ext_vector_type(16))) __bf16       v16bf;
typedef __attribute__((ext_vector_type(8)))  float        v8f;
typedef __attribute__((ext_vector_type(8)))  unsigned int v8u;
typedef __attribute__((ext_vector_type(4)))  unsigned int v4u;
typedef __attribute__((ext_vector_type(4)))  float        v4f;

// round-to-nearest-even fp32 -> packed 2x bf16
__device__ __forceinline__ unsigned pk_bf16(float a, float b) {
  unsigned ua = __float_as_uint(a), ub = __float_as_uint(b);
  ua += 0x7FFFu + ((ua >> 16) & 1u);
  ub += 0x7FFFu + ((ub >> 16) & 1u);
  return (ua >> 16) | (ub & 0xFFFF0000u);
}
__device__ __forceinline__ unsigned short bf16_1(float a) {
  unsigned ua = __float_as_uint(a);
  ua += 0x7FFFu + ((ua >> 16) & 1u);
  return (unsigned short)(ua >> 16);
}
__device__ __forceinline__ float sigmoid_(float x) {
  return 1.0f / (1.0f + __expf(-x));
}

// Device-wide monotonic barrier (counter zeroed by init kernel each launch).
__device__ __forceinline__ void gbar(unsigned* bar, unsigned target, int tid) {
  __syncthreads();
  if (tid == 0) {
    __hip_atomic_fetch_add(bar, 1u, __ATOMIC_RELEASE, __HIP_MEMORY_SCOPE_AGENT);
    while (__hip_atomic_load(bar, __ATOMIC_ACQUIRE, __HIP_MEMORY_SCOPE_AGENT) < target) {
      __builtin_amdgcn_s_sleep(1);
    }
  }
  __syncthreads();
}

__global__ void init_bar(unsigned* bar) {
  if (threadIdx.x == 0) *bar = 0u;
}

// Parallel pre-pass: ins fp32 -> bf16 (row-major), off the scan critical path.
// 8 elements per thread: 2x b128 load -> 1x b128 store.
__global__ void __launch_bounds__(256)
conv_x(const float* __restrict__ ins, unsigned short* __restrict__ xbf) {
  size_t i = ((size_t)blockIdx.x * 256 + threadIdx.x) * 8;
  const v4f* p = (const v4f*)(ins + i);
  v4f f0 = p[0], f1 = p[1];
  v4u o;
  o[0] = pk_bf16(f0[0], f0[1]);
  o[1] = pk_bf16(f0[2], f0[3]);
  o[2] = pk_bf16(f1[0], f1[1]);
  o[3] = pk_bf16(f1[2], f1[3]);
  *(v4u*)(xbf + i) = o;
}

template <bool PRECONV>
__global__ void __launch_bounds__(THREADS, 1)
gru_scan(const float* __restrict__ ins, const unsigned short* __restrict__ xbf,
         const float* __restrict__ h0,
         const float* __restrict__ Wi, const float* __restrict__ bi,
         const float* __restrict__ Whrz, const float* __restrict__ Whn,
         const float* __restrict__ bhn,
         float* __restrict__ out, unsigned* __restrict__ bar,
         unsigned short* __restrict__ hbuf0, unsigned short* __restrict__ hbuf1)
{
  // Weight strips in B-operand WMMA layout, bf16 packed as uints:
  // 48 k-steps: [0..15]=r (K=512: x then h), [16..31]=z, [32..39]=n_in (x), [40..47]=n_h (h).
  // Per k-step: 32 lanes * 8 uints (two 16B chunks -> VGPRs 0-3 and 4-7).
  __shared__ unsigned ldsW[48 * 256];          // 48 KB
  __shared__ float    ldsH[BATCH * 16];        // 16 KB: own fp32 h slice

  const int tid  = threadIdx.x;
  const int s    = blockIdx.x;                 // H slice 0..15
  const int lane = tid & 31;
  const int w    = tid >> 5;                   // wave id == M tile
  const int n    = lane & 15;
  const int half = lane >> 4;
  const int j    = 16 * s + n;                 // this lane's global H column

  // ---------------- prepack weights (fp32 -> bf16, B layout) ----------------
  for (int u = tid; u < 48 * 256; u += THREADS) {
    int ks = u >> 8;           // absolute k-step 0..47
    int r  = u & 255;
    int pl = r >> 3;           // owning lane
    int q  = r & 7;            // uint within lane's 8
    int chunk = q >> 2;        // 0: VGPR0-3, 1: VGPR4-7
    int i  = q & 3;
    int pn = pl & 15, ph = pl >> 4;
    int koff = chunk * 16 + ph * 8 + 2 * i;
    int col  = 16 * s + pn;
    int tile, kt;
    if (ks < 16)      { tile = 0; kt = ks; }
    else if (ks < 32) { tile = 1; kt = ks - 16; }
    else if (ks < 40) { tile = 2; kt = ks - 32; }
    else              { tile = 3; kt = ks - 40; }
    int k = kt * 32 + koff;
    float w0, w1;
    if (tile == 0) {           // r: x part then h part
      if (k < 256) { w0 = Wi[k * 768 + col];            w1 = Wi[(k + 1) * 768 + col]; }
      else { int kk = k - 256; w0 = Whrz[kk * 512 + col]; w1 = Whrz[(kk + 1) * 512 + col]; }
    } else if (tile == 1) {    // z
      if (k < 256) { w0 = Wi[k * 768 + 256 + col];      w1 = Wi[(k + 1) * 768 + 256 + col]; }
      else { int kk = k - 256; w0 = Whrz[kk * 512 + 256 + col]; w1 = Whrz[(kk + 1) * 512 + 256 + col]; }
    } else if (tile == 2) {    // n input part
      w0 = Wi[k * 768 + 512 + col]; w1 = Wi[(k + 1) * 768 + 512 + col];
    } else {                   // n hidden part
      w0 = Whn[k * 256 + col];      w1 = Whn[(k + 1) * 256 + col];
    }
    ldsW[u] = pk_bf16(w0, w1);
  }

  // biases (per N column; broadcast over rows)
  const float bir = bi[j], biz = bi[256 + j], bin = bi[512 + j], bh = bhn[j];

  // init own h slice (fp32 in LDS) + publish bf16 h to buffer 0
  for (int e = tid; e < BATCH * 16; e += THREADS) {
    int b = e >> 4, c = e & 15;
    float v = h0[b * HID + 16 * s + c];
    ldsH[e] = v;
    hbuf0[b * HID + 16 * s + c] = bf16_1(v);
  }

  __threadfence();
  gbar(bar, NWG, tid);
  __builtin_amdgcn_fence(__ATOMIC_ACQUIRE, "agent");

  const int mBase = 16 * w;
  const int mRow  = mBase + (lane & 15);       // row this lane supplies for A operand

  for (int t = 0; t < T_STEPS; ++t) {
    const unsigned short* hrd = (t & 1) ? hbuf1 : hbuf0;
    unsigned short*       hwr = (t & 1) ? hbuf0 : hbuf1;

    v8f accR = {}, accZ = {}, accNI = {}, accNH = {};

    const float* xrowf = ins + ((size_t)t * BATCH + mRow) * DIM;
    const v4u*   xrowb = (const v4u*)(xbf + ((size_t)t * BATCH + mRow) * DIM);
    const v4u*   hrow  = (const v4u*)(hrd + (size_t)mRow * HID);

    if (t + 1 < T_STEPS) {  // warm L2 for next timestep's x tile
      if (PRECONV)
        __builtin_prefetch(xbf + ((size_t)(t + 1) * BATCH + mRow) * DIM + half * 128, 0, 0);
      else
        __builtin_prefetch(ins + ((size_t)(t + 1) * BATCH + mRow) * DIM + half * 128, 0, 0);
    }

    auto ldB = [&](int ksAbs) -> v16bf {
      const v4u* p = (const v4u*)&ldsW[ksAbs * 256 + lane * 8];
      v4u c0 = p[0], c1 = p[1];
      v8u b;
      b[0] = c0[0]; b[1] = c0[1]; b[2] = c0[2]; b[3] = c0[3];
      b[4] = c1[0]; b[5] = c1[1]; b[6] = c1[2]; b[7] = c1[3];
      return __builtin_bit_cast(v16bf, b);
    };

#pragma unroll
    for (int ks = 0; ks < 8; ++ks) {
      // ---- A = x_t tile ----
      v16bf Ax;
      if (PRECONV) {
        v4u x0 = xrowb[ks * 4 + half];
        v4u x1 = xrowb[ks * 4 + 2 + half];
        v8u ax;
        ax[0] = x0[0]; ax[1] = x0[1]; ax[2] = x0[2]; ax[3] = x0[3];
        ax[4] = x1[0]; ax[5] = x1[1]; ax[6] = x1[2]; ax[7] = x1[3];
        Ax = __builtin_bit_cast(v16bf, ax);
      } else {
        const v4f* xp = (const v4f*)(xrowf + ks * 32 + half * 8);
        v4f f0 = xp[0], f1 = xp[1];     // K = base .. base+7
        v4f f2 = xp[4], f3 = xp[5];     // K = base+16 .. base+23
        v8u ax;
        ax[0] = pk_bf16(f0[0], f0[1]); ax[1] = pk_bf16(f0[2], f0[3]);
        ax[2] = pk_bf16(f1[0], f1[1]); ax[3] = pk_bf16(f1[2], f1[3]);
        ax[4] = pk_bf16(f2[0], f2[1]); ax[5] = pk_bf16(f2[2], f2[3]);
        ax[6] = pk_bf16(f3[0], f3[1]); ax[7] = pk_bf16(f3[2], f3[3]);
        Ax = __builtin_bit_cast(v16bf, ax);
      }

      accR  = __builtin_amdgcn_wmma_f32_16x16x32_bf16(false, Ax, false, ldB(ks),      (short)0, accR,  false, false);
      accZ  = __builtin_amdgcn_wmma_f32_16x16x32_bf16(false, Ax, false, ldB(16 + ks), (short)0, accZ,  false, false);
      accNI = __builtin_amdgcn_wmma_f32_16x16x32_bf16(false, Ax, false, ldB(32 + ks), (short)0, accNI, false, false);

      // ---- A = h tile (bf16, row-major, 2x b128 per lane) ----
      v4u hc0 = hrow[ks * 4 + half];
      v4u hc1 = hrow[ks * 4 + 2 + half];
      v8u ah;
      ah[0] = hc0[0]; ah[1] = hc0[1]; ah[2] = hc0[2]; ah[3] = hc0[3];
      ah[4] = hc1[0]; ah[5] = hc1[1]; ah[6] = hc1[2]; ah[7] = hc1[3];
      v16bf Ah = __builtin_bit_cast(v16bf, ah);

      accR  = __builtin_amdgcn_wmma_f32_16x16x32_bf16(false, Ah, false, ldB(8 + ks),  (short)0, accR,  false, false);
      accZ  = __builtin_amdgcn_wmma_f32_16x16x32_bf16(false, Ah, false, ldB(24 + ks), (short)0, accZ,  false, false);
      accNH = __builtin_amdgcn_wmma_f32_16x16x32_bf16(false, Ah, false, ldB(40 + ks), (short)0, accNH, false, false);
    }

    // ---- elementwise GRU update; C layout: VGPR v -> rows v / v+8, lane -> col ----
    float* outy = out + (size_t)BATCH * HID + (size_t)t * BATCH * HID;
#pragma unroll
    for (int v = 0; v < 8; ++v) {
      int b = mBase + v + 8 * half;
      float rr = sigmoid_(accR[v] + bir);
      float zz = sigmoid_(accZ[v] + biz);
      float nn = tanhf(accNI[v] + bin + rr * (accNH[v] + bh));
      float hp = ldsH[b * 16 + n];
      float hnew = (1.0f - zz) * nn + zz * hp;
      ldsH[b * 16 + n] = hnew;                    // element owned by this thread forever
      outy[(size_t)b * HID + j] = hnew;           // ys[t]
      hwr[b * HID + j] = bf16_1(hnew);            // publish for next step's A operand
    }

    __threadfence();
    gbar(bar, (unsigned)((t + 2) * NWG), tid);
    __builtin_amdgcn_fence(__ATOMIC_ACQUIRE, "agent");
  }

  // final carry (h after last step)
  for (int e = tid; e < BATCH * 16; e += THREADS) {
    int b = e >> 4, c = e & 15;
    out[b * HID + 16 * s + c] = ldsH[e];
  }
}

extern "C" void kernel_launch(void* const* d_in, const int* in_sizes, int n_in,
                              void* d_out, int out_size, void* d_ws, size_t ws_size,
                              hipStream_t stream) {
  const float* ins  = (const float*)d_in[0];
  // d_in[1] = resets: unused by the reference module
  const float* h0   = (const float*)d_in[2];
  const float* Wi   = (const float*)d_in[3];
  const float* bi   = (const float*)d_in[4];
  const float* Whrz = (const float*)d_in[5];
  const float* Whn  = (const float*)d_in[6];
  const float* bhn  = (const float*)d_in[7];
  float* out = (float*)d_out;

  const size_t HB      = (size_t)BATCH * HID * 2;              // bf16 h buffer bytes
  const size_t XBF_OFF = 1024 + 2 * HB;                        // after bar + 2 h buffers
  const size_t XBF_BYTES = (size_t)T_STEPS * BATCH * DIM * 2;  // 134 MB bf16 copy of ins

  unsigned*       barp  = (unsigned*)d_ws;
  unsigned short* hbuf0 = (unsigned short*)((char*)d_ws + 1024);
  unsigned short* hbuf1 = (unsigned short*)((char*)d_ws + 1024 + HB);
  unsigned short* xbf   = (unsigned short*)((char*)d_ws + XBF_OFF);

  init_bar<<<1, 32, 0, stream>>>(barp);

  // Deterministic host-side choice: ws_size is fixed across calls.
  if (ws_size >= XBF_OFF + XBF_BYTES) {
    // Hoist x fp32->bf16 conversion off the scan's sequential critical path.
    const size_t nElem = (size_t)T_STEPS * BATCH * DIM;        // 2^26
    const int blocks = (int)(nElem / (256 * 8));
    conv_x<<<blocks, 256, 0, stream>>>(ins, xbf);
    gru_scan<true><<<NWG, THREADS, 0, stream>>>(ins, xbf, h0, Wi, bi, Whrz, Whn, bhn,
                                                out, barp, hbuf0, hbuf1);
  } else {
    gru_scan<false><<<NWG, THREADS, 0, stream>>>(ins, xbf, h0, Wi, bi, Whrz, Whn, bhn,
                                                 out, barp, hbuf0, hbuf1);
  }
}
